// LocalAttentionBlock_38680475468385
// MI455X (gfx1250) — matrix operand order repaired and checked by reading, not compiled, not run on hardware
//
#include <hip/hip_runtime.h>
#include <hip/hip_bf16.h>

// ---------------------------------------------------------------------------
// LocalAttentionBlock on MI455X (gfx1250, wave32):
//   h1 = n1_w*asinh(h)+n1_b                       [elementwise, f32->bf16]
//   qkv = h1 @ qkv_w + qkv_b  (q *= hd^-0.5)      [WMMA bf16 GEMM, f32 out]
//   attn = NATTEN 3x3 softmax attention           [VALU, wave32 shuffles]
//   h    = h + attn @ proj_w + proj_b             [WMMA GEMM, fused residual]
//   h2   = n2_w*asinh(h)+n2_b                     [fused into proj epilogue]
//   mid  = gelu(h2 @ mlp_w1 + mlp_b1)             [WMMA GEMM, fused gelu]
//   out  = h + mid @ mlp_w2 + mlp_b2              [WMMA GEMM, fused residual]
//
// GEMM: 128x128 tile / workgroup (8 waves, wave = 16 rows x 128 cols),
// K-step 32, double-buffered LDS fed by GLOBAL_LOAD_ASYNC_TO_LDS_B128
// (ASYNCcnt-tracked DMA, no VGPR round-trip), weights pre-transposed to
// [N][K] so both tiles stage as contiguous 16B chunks.
// ---------------------------------------------------------------------------

typedef __attribute__((ext_vector_type(16))) __bf16        v16bf;
typedef __attribute__((ext_vector_type(8)))  float         v8f;
typedef __attribute__((ext_vector_type(4)))  unsigned int  u32x4;

union FragU { u32x4 u[2]; v16bf v; };   // 32 bytes: one WMMA bf16 fragment per lane

#define NTOK   65536      // B*H*W = 16*64*64
#define CDIM   256
#define HD     32
#define QSCALE 0.17677669529663689f   // 32^-0.5

// --- CDNA5 async global->LDS copy (GVS mode: saddr base + 32-bit vgpr offset)
__device__ __forceinline__ unsigned lds_off32(const void* p) {
  // flat->LDS translation uses addr[31:0] (ISA 10.2): truncate generic pointer
  return (unsigned)(unsigned long long)p;
}
__device__ __forceinline__ void async_ld16(unsigned lds, unsigned long long base,
                                           unsigned byte_off) {
  asm volatile("global_load_async_to_lds_b128 %0, %1, %2"
               : : "v"(lds), "v"(byte_off), "s"(base) : "memory");
}
__device__ __forceinline__ void wait_async0() {
  asm volatile("s_wait_asynccnt 0x0" : : : "memory");
}

// ---------------------------------------------------------------------------
// Cast f32 weight [K][N] -> bf16 transposed [N][K]
__global__ __launch_bounds__(256) void cast_t_bf16_kernel(
    const float* __restrict__ in, __hip_bfloat16* __restrict__ out, int K, int N) {
  int i = blockIdx.x * 256 + threadIdx.x;
  if (i < K * N) {
    int k = i / N, n = i - k * N;
    out[(size_t)n * K + k] = __float2bfloat16(in[i]);
  }
}

// Elementwise: h1 = n1_w*asinh(h)+n1_b, stored bf16
__global__ __launch_bounds__(256) void norm_bf16_kernel(
    const float* __restrict__ h, const float* __restrict__ w,
    const float* __restrict__ b, __hip_bfloat16* __restrict__ out, int n) {
  int i = blockIdx.x * 256 + threadIdx.x;
  if (i < n) {
    int c = i & (CDIM - 1);
    out[i] = __float2bfloat16(w[c] * asinhf(h[i]) + b[c]);
  }
}

// ---------------------------------------------------------------------------
// NATTEN 3x3 attention. One block per token; wave w = head w; lane = channel.
// qkv layout: [tok][768] = [q(256) | k(256) | v(256)], q pre-scaled.
__global__ __launch_bounds__(256) void natten3x3_kernel(
    const float* __restrict__ qkv, __hip_bfloat16* __restrict__ out) {
  const int tok  = blockIdx.x;
  const int head = threadIdx.x >> 5;
  const int d    = threadIdx.x & 31;
  const int b = tok >> 12, i = (tok >> 6) & 63, j = tok & 63;
  int ri = i - 1; ri = ri < 0 ? 0 : (ri > 61 ? 61 : ri);
  int cj = j - 1; cj = cj < 0 ? 0 : (cj > 61 ? 61 : cj);

  const float q = qkv[(size_t)tok * 768 + head * HD + d];
  float sc[9];
#pragma unroll
  for (int a = 0; a < 3; ++a)
#pragma unroll
    for (int c = 0; c < 3; ++c) {
      int nt = (b << 12) | ((ri + a) << 6) | (cj + c);
      float kv = qkv[(size_t)nt * 768 + CDIM + head * HD + d];
      float p = q * kv;
#pragma unroll
      for (int off = 16; off > 0; off >>= 1) p += __shfl_xor(p, off, 32);
      sc[a * 3 + c] = p;             // full reduction -> replicated on all lanes
    }
  float mx = sc[0];
#pragma unroll
  for (int m = 1; m < 9; ++m) mx = fmaxf(mx, sc[m]);
  float s = 0.f;
#pragma unroll
  for (int m = 0; m < 9; ++m) { sc[m] = __expf(sc[m] - mx); s += sc[m]; }
  const float inv = 1.f / s;
  float o = 0.f;
#pragma unroll
  for (int a = 0; a < 3; ++a)
#pragma unroll
    for (int c = 0; c < 3; ++c) {
      int nt = (b << 12) | ((ri + a) << 6) | (cj + c);
      float vv = qkv[(size_t)nt * 768 + 2 * CDIM + head * HD + d];
      o = fmaf(sc[a * 3 + c] * inv, vv, o);
    }
  out[(size_t)tok * CDIM + head * HD + d] = __float2bfloat16(o);
}

// ---------------------------------------------------------------------------
// Tiled bf16 WMMA GEMM: C[M,N] = A[M,K] @ Bt[N,K]^T (+ fused epilogue).
// Double-buffered LDS, async-DMA pipeline, batched wmma issue.
enum { EP_QKV = 0, EP_PROJ = 1, EP_GELU = 2, EP_MLP2 = 3 };

template <int EP>
__global__ __launch_bounds__(256) void gemm_wmma_bf16_kernel(
    const __hip_bfloat16* __restrict__ A,   // [M][K] bf16
    const __hip_bfloat16* __restrict__ Bt,  // [N][K] bf16 (transposed weight)
    int K, int N,
    const float* __restrict__ bias,         // [N]
    const float* __restrict__ res,          // residual [M][N] (PROJ/MLP2)
    float* __restrict__ outF,               // f32 out (QKV/PROJ/MLP2)
    __hip_bfloat16* __restrict__ outB,      // bf16 out (PROJ h2 / GELU mid)
    const float* __restrict__ nw, const float* __restrict__ nb) {
  __shared__ __align__(16) unsigned short lA[2][128 * 32];
  __shared__ __align__(16) unsigned short lB[2][128 * 32];   // [n][k]

  const int tid = threadIdx.x;
  const int wid = tid >> 5, lane = tid & 31;
  const int ln = lane & 15, lh = lane >> 4;
  const int n0 = blockIdx.x * 128, m0 = blockIdx.y * 128;

  // staging assignment: thread covers rows srow and srow+64, 8 bf16 at sseg
  const int srow = tid >> 2, sseg = (tid & 3) * 8;
  const unsigned long long baseA = (unsigned long long)A;
  const unsigned long long baseB = (unsigned long long)Bt;
  const unsigned offA0 = (unsigned)(((size_t)(m0 + srow) * K + sseg) * 2);
  const unsigned offA1 = offA0 + (unsigned)((size_t)64 * K * 2);
  const unsigned offB0 = (unsigned)(((size_t)(n0 + srow) * K + sseg) * 2);
  const unsigned offB1 = offB0 + (unsigned)((size_t)64 * K * 2);
  const unsigned ldsA0 = lds_off32(&lA[0][srow * 32 + sseg]);
  const unsigned ldsA1 = lds_off32(&lA[0][(srow + 64) * 32 + sseg]);
  const unsigned ldsB0 = lds_off32(&lB[0][srow * 32 + sseg]);
  const unsigned ldsB1 = lds_off32(&lB[0][(srow + 64) * 32 + sseg]);
  const unsigned bufB = 128 * 32 * 2;   // bytes per LDS buffer

  v8f acc[8];
#pragma unroll
  for (int t = 0; t < 8; ++t)
#pragma unroll
    for (int j = 0; j < 8; ++j) acc[t][j] = 0.0f;

  // ---- prologue: async-stage K-step 0 into buffer 0
  async_ld16(ldsA0, baseA, offA0);
  async_ld16(ldsA1, baseA, offA1);
  async_ld16(ldsB0, baseB, offB0);
  async_ld16(ldsB1, baseB, offB1);
  wait_async0();
  __syncthreads();

  const int ksteps = K >> 5;
  for (int kt = 0; kt < ksteps; ++kt) {
    const int cur = kt & 1;
    const bool more = (kt + 1) < ksteps;
    // ---- kick off DMA for the next K-step into the alternate buffer
    if (more) {
      const unsigned kb  = (unsigned)(kt + 1) << 6;   // 32 bf16 = 64 bytes
      const unsigned nb_ = (unsigned)(cur ^ 1) * bufB;
      async_ld16(ldsA0 + nb_, baseA, offA0 + kb);
      async_ld16(ldsA1 + nb_, baseA, offA1 + kb);
      async_ld16(ldsB0 + nb_, baseB, offB0 + kb);
      async_ld16(ldsB1 + nb_, baseB, offB1 + kb);
    }

    // ---- A fragment: row = wid*16+ln ; VGPR0-3: K = lh*8+0..7 ; VGPR4-7: +16
    FragU fa;
    const unsigned short* pa = &lA[cur][(wid * 16 + ln) * 32 + lh * 8];
    fa.u[0] = *(const u32x4*)pa;
    fa.u[1] = *(const u32x4*)(pa + 16);
    // ---- 8 B fragments + 8 wmma (compute overlaps the async DMA above)
    FragU fb[8];
#pragma unroll
    for (int t = 0; t < 8; ++t) {
      const unsigned short* pb = &lB[cur][(t * 16 + ln) * 32 + lh * 8];
      fb[t].u[0] = *(const u32x4*)pb;
      fb[t].u[1] = *(const u32x4*)(pb + 16);
    }
#pragma unroll
    for (int t = 0; t < 8; ++t)
      acc[t] = __builtin_amdgcn_wmma_f32_16x16x32_bf16(
          false, fa.v, false, fb[t].v, (short)0, acc[t], false, false);

    if (more) wait_async0();   // own DMA done; barrier covers other waves
    __syncthreads();           // reads of `cur` done; `cur^1` fully resident
  }

  // ---- epilogue. C/D layout: lane ln = col, VGPR j -> row j + 8*lh.
  const int mrow = m0 + wid * 16 + lh * 8;
#pragma unroll
  for (int t = 0; t < 8; ++t) {
    const int col = n0 + t * 16 + ln;
    const float bv = bias[col];
#pragma unroll
    for (int j = 0; j < 8; ++j) {
      const int m = mrow + j;
      float v = acc[t][j] + bv;
      if constexpr (EP == EP_QKV) {
        if (col < CDIM) v *= QSCALE;            // scale q by hd^-0.5
        outF[(size_t)m * N + col] = v;
      } else if constexpr (EP == EP_PROJ) {
        v += res[(size_t)m * N + col];          // residual h
        outF[(size_t)m * N + col] = v;          // h_new (f32, for final residual)
        float t2 = nw[col] * asinhf(v) + nb[col];
        outB[(size_t)m * N + col] = __float2bfloat16(t2);   // h2 bf16
      } else if constexpr (EP == EP_GELU) {
        float g = 0.5f * v * (1.0f + erff(v * 0.70710678118654752f));
        outB[(size_t)m * N + col] = __float2bfloat16(g);
      } else {  // EP_MLP2
        v += res[(size_t)m * N + col];          // residual h_new
        outF[(size_t)m * N + col] = v;          // final output
      }
    }
  }
}

// ---------------------------------------------------------------------------
extern "C" void kernel_launch(void* const* d_in, const int* in_sizes, int n_in,
                              void* d_out, int out_size, void* d_ws, size_t ws_size,
                              hipStream_t stream) {
  const float* h      = (const float*)d_in[0];
  const float* qkv_w  = (const float*)d_in[1];
  const float* qkv_b  = (const float*)d_in[2];
  const float* proj_w = (const float*)d_in[3];
  const float* proj_b = (const float*)d_in[4];
  const float* n1_w   = (const float*)d_in[5];
  const float* n1_b   = (const float*)d_in[6];
  const float* n2_w   = (const float*)d_in[7];
  const float* n2_b   = (const float*)d_in[8];
  const float* mlp_w1 = (const float*)d_in[9];
  const float* mlp_b1 = (const float*)d_in[10];
  const float* mlp_w2 = (const float*)d_in[11];
  const float* mlp_b2 = (const float*)d_in[12];
  float* dout = (float*)d_out;

  // --- workspace layout (256B aligned); buffers reused across phases
  char* ws = (char*)d_ws;
  size_t off = 0;
  auto alloc = [&](size_t bytes) {
    char* p = ws + off;
    off += (bytes + 255) & ~(size_t)255;
    return p;
  };
  __hip_bfloat16* w_qkvT  = (__hip_bfloat16*)alloc((size_t)CDIM * 3 * CDIM * 2); // [768][256]
  __hip_bfloat16* w_projT = (__hip_bfloat16*)alloc((size_t)CDIM * CDIM * 2);     // [256][256]
  __hip_bfloat16* w_m1T   = (__hip_bfloat16*)alloc((size_t)CDIM * 2 * CDIM * 2); // [512][256]
  __hip_bfloat16* w_m2T   = (__hip_bfloat16*)alloc((size_t)2 * CDIM * CDIM * 2); // [256][512]
  __hip_bfloat16* h1h2    = (__hip_bfloat16*)alloc((size_t)NTOK * CDIM * 2);     // h1, later h2
  float*          qkvmid  = (float*)alloc((size_t)NTOK * 3 * CDIM * 4);          // qkv, later mid(bf16)
  __hip_bfloat16* attnb   = (__hip_bfloat16*)alloc((size_t)NTOK * CDIM * 2);
  float*          hnew    = (float*)alloc((size_t)NTOK * CDIM * 4);
  __hip_bfloat16* midb    = (__hip_bfloat16*)qkvmid;                             // reuse (qkv dead)

  // --- 1. weights -> bf16, transposed to [N][K]
  cast_t_bf16_kernel<<<(CDIM * 3 * CDIM + 255) / 256, 256, 0, stream>>>(qkv_w,  w_qkvT,  CDIM, 3 * CDIM);
  cast_t_bf16_kernel<<<(CDIM * CDIM + 255) / 256, 256, 0, stream>>>(proj_w, w_projT, CDIM, CDIM);
  cast_t_bf16_kernel<<<(CDIM * 2 * CDIM + 255) / 256, 256, 0, stream>>>(mlp_w1, w_m1T, CDIM, 2 * CDIM);
  cast_t_bf16_kernel<<<(2 * CDIM * CDIM + 255) / 256, 256, 0, stream>>>(mlp_w2, w_m2T, 2 * CDIM, CDIM);

  // --- 2. h1 = arsinh-norm(h) -> bf16
  norm_bf16_kernel<<<(NTOK * CDIM) / 256, 256, 0, stream>>>(h, n1_w, n1_b, h1h2, NTOK * CDIM);

  // --- 3. qkv = h1 @ qkv_w + b (q scaled); [65536,256]x[256,768]
  gemm_wmma_bf16_kernel<EP_QKV><<<dim3(768 / 128, NTOK / 128), 256, 0, stream>>>(
      h1h2, w_qkvT, CDIM, 3 * CDIM, qkv_b, nullptr, qkvmid, nullptr, nullptr, nullptr);

  // --- 4. NATTEN 3x3 attention -> bf16
  natten3x3_kernel<<<NTOK, 256, 0, stream>>>(qkvmid, attnb);

  // --- 5. h_new = h + attn @ proj_w + b ; h2 = arsinh-norm(h_new) (fused)
  gemm_wmma_bf16_kernel<EP_PROJ><<<dim3(CDIM / 128, NTOK / 128), 256, 0, stream>>>(
      attnb, w_projT, CDIM, CDIM, proj_b, h, hnew, h1h2, n2_w, n2_b);

  // --- 6. mid = gelu(h2 @ mlp_w1 + b1) -> bf16 (reuses qkv buffer)
  gemm_wmma_bf16_kernel<EP_GELU><<<dim3(2 * CDIM / 128, NTOK / 128), 256, 0, stream>>>(
      h1h2, w_m1T, CDIM, 2 * CDIM, mlp_b1, nullptr, nullptr, midb, nullptr, nullptr);

  // --- 7. out = h_new + mid @ mlp_w2 + b2
  gemm_wmma_bf16_kernel<EP_MLP2><<<dim3(CDIM / 128, NTOK / 128), 256, 0, stream>>>(
      midb, w_m2T, 2 * CDIM, CDIM, mlp_b2, hnew, dout, nullptr, nullptr, nullptr);

  (void)in_sizes; (void)n_in; (void)out_size; (void)ws_size;
}
